// WindowAttention_25486335934873
// MI455X (gfx1250) — compile-verified, matrix operand order checked by reading, not run on hardware
//
#include <hip/hip_runtime.h>
#include <hip/hip_bf16.h>

typedef __attribute__((ext_vector_type(16))) _Float16 v16h;
typedef __attribute__((ext_vector_type(8)))  _Float16 v8h;
typedef __attribute__((ext_vector_type(4)))  _Float16 v4h;
typedef __attribute__((ext_vector_type(8)))  float    v8f;

#define NTOK 49
#define DIM 192
#define HEADS 6
#define NUM_WIN 1024
#define SCALE 0.17677669529663687f   // 1/sqrt(32)

// LDS strides (elements); all f16 row strides are 16-byte aligned
#define XS  200   // halves: row stride for X/Q/K/O (64 x 192), 400 B
#define VTS 72    // halves: row stride for Vt (192 x 64), 144 B
#define SST 65    // floats: row stride for S (64 x 64 per head)
#define PST 72    // halves: row stride for P (64 x 64 per head), 144 B

// LDS byte offsets (S overlays X, which is dead once Q/K/Vt exist)
#define OFF_Q  0                    // 25600
#define OFF_K  25600                // 25600
#define OFF_VT 51200                // 192*72*2 = 27648
#define OFF_X  78848                // 25600
#define OFF_S  78848                // 6*64*65*4 = 99840 (overlays X)
#define OFF_P  178688               // 6*64*72*2 = 55296
#define OFF_O  233984               // 25600
#define LDS_BYTES 259584

__device__ __forceinline__ v8f wmma16(v16h a, v16h b, v8f c) {
    return __builtin_amdgcn_wmma_f32_16x16x32_f16(false, a, false, b, (short)0, c,
                                                  false, false);
}

// LDS fragment: lane's 16 halves live at p[0..7] and p[16..23] (both 16B aligned)
__device__ __forceinline__ v16h frag_lds(const _Float16* p) {
    v8h lo = *(const v8h*)p;
    v8h hi = *(const v8h*)(p + 16);
    return __builtin_shufflevector(lo, hi, 0, 1, 2, 3, 4, 5, 6, 7,
                                           8, 9, 10, 11, 12, 13, 14, 15);
}

// Pre-swizzled (fragment-major) global weights: 16 contiguous halves per lane
__device__ __forceinline__ v16h frag_glb(const _Float16* p) {
    v8h lo = *(const v8h*)p;
    v8h hi = *(const v8h*)(p + 8);
    return __builtin_shufflevector(lo, hi, 0, 1, 2, 3, 4, 5, 6, 7,
                                           8, 9, 10, 11, 12, 13, 14, 15);
}

// ---------------------------------------------------------------------------
// Kernel 1a: convert + swizzle the 4 weight matrices into fragment-major f16.
// ws halves layout: idx = (((mat*6 + kt)*12 + nt)*32 + lane)*16 + j
//   j<8  -> K = kt*32 + kb + j        (kb = lane&16 ? 8 : 0)
//   j>=8 -> K = kt*32 + kb + 16 + (j-8)
//   N    = nt*16 + (lane&15)
__global__ __launch_bounds__(256)
void wa_convert_weights(const float* __restrict__ Wq, const float* __restrict__ Wk,
                        const float* __restrict__ Wv, const float* __restrict__ Wp,
                        _Float16* __restrict__ wts) {
    int idx = blockIdx.x * 256 + threadIdx.x;
    if (idx >= 4 * DIM * DIM) return;
    int j    = idx & 15;
    int lane = (idx >> 4) & 31;
    int nt   = (idx >> 9) % 12;
    int ktm  = (idx >> 9) / 12;       // mat*6 + kt
    int kt   = ktm % 6;
    int mat  = ktm / 6;
    int kb   = (lane & 16) ? 8 : 0;
    int k    = kt * 32 + kb + ((j < 8) ? j : j + 8);
    int n    = nt * 16 + (lane & 15);
    const float* src = (mat == 0) ? Wq : (mat == 1) ? Wk : (mat == 2) ? Wv : Wp;
    float v = src[k * DIM + n];
    if (mat == 0) v *= SCALE;
    wts[idx] = (_Float16)v;
}

// Kernel 1b: pre-gather relative position bias -> biasP[h][n][m], f32.
__global__ __launch_bounds__(256)
void wa_build_bias(const float* __restrict__ rel_bias, const int* __restrict__ pos_index,
                   float* __restrict__ biasP) {
    int idx = blockIdx.x * 256 + threadIdx.x;
    if (idx >= HEADS * NTOK * NTOK) return;
    int h = idx / (NTOK * NTOK), nm = idx % (NTOK * NTOK);
    biasP[idx] = rel_bias[pos_index[nm] * HEADS + h];
}

// ---------------------------------------------------------------------------
// Kernel 2: fully fused window attention.  One workgroup (8 wave32) per window.
__global__ __launch_bounds__(256)
void wa_attn_kernel(const float* __restrict__ x, const float* __restrict__ mask,
                    const float* __restrict__ bq, const float* __restrict__ bk,
                    const float* __restrict__ bv, const float* __restrict__ bp,
                    const _Float16* __restrict__ wts, const float* __restrict__ biasP,
                    float* __restrict__ out) {
    extern __shared__ char smem[];
    _Float16* Qh  = (_Float16*)(smem + OFF_Q);
    _Float16* Kh  = (_Float16*)(smem + OFF_K);
    _Float16* Vt  = (_Float16*)(smem + OFF_VT);  // transposed: Vt[d][m]
    _Float16* Xh  = (_Float16*)(smem + OFF_X);
    float*    Sf  = (float*)   (smem + OFF_S);
    _Float16* Ph  = (_Float16*)(smem + OFF_P);
    _Float16* Oh  = (_Float16*)(smem + OFF_O);

    const int tid  = threadIdx.x;
    const int wave = tid >> 5;
    const int lane = tid & 31;
    const int l15  = lane & 15;
    const int kb   = (lane & 16) ? 8 : 0;   // per-lane K base within 32-wide slab
    const int hi8  = (lane & 16) ? 8 : 0;   // per-lane M offset in C/D tiles
    const int blk  = blockIdx.x;
    const int win  = blk & (NUM_WIN - 1);

    // ---- stage 1: X (49x192 f32) -> LDS f16 via float4 loads / 4-half stores
    {
        const float* xb = x + (size_t)blk * NTOK * DIM;
        for (int idx = tid; idx < 64 * (DIM / 4); idx += 256) {
            int r = idx / (DIM / 4), c4 = idx % (DIM / 4);
            v4h hv;
            if (r < NTOK) {
                float4 f = *(const float4*)(xb + r * DIM + c4 * 4);
                hv[0] = (_Float16)f.x; hv[1] = (_Float16)f.y;
                hv[2] = (_Float16)f.z; hv[3] = (_Float16)f.w;
            } else {
                hv[0] = hv[1] = hv[2] = hv[3] = (_Float16)0.0f;
            }
            *(v4h*)(Xh + r * XS + c4 * 4) = hv;
        }
    }
    __syncthreads();

    // ---- stage 2: Q,K,V = X @ W + b  (3 mats * 4x12 tiles; K=192 -> 6 wmma)
    //      V is stored transposed (Vt[d][m]) so stage 5's B operand is row-major
    for (int t = wave; t < 144; t += 8) {
        int mat = t % 3;
        int tt  = t / 3;
        int mt  = tt / 12, nt = tt % 12;
        int colg = nt * 16 + l15;
        const float* bias = (mat == 0) ? bq : (mat == 1) ? bk : bv;
        float b0 = bias[colg];
        if (mat == 0) b0 *= SCALE;          // q is pre-scaled (Wq too)
        v8f acc;
#pragma unroll
        for (int r = 0; r < 8; ++r) acc[r] = b0;
        int arow = mt * 16 + l15;
#pragma unroll
        for (int kt = 0; kt < 6; ++kt) {
            v16h a = frag_lds(Xh + arow * XS + kt * 32 + kb);
            v16h b = frag_glb(wts + (((mat * 6 + kt) * 12 + nt) * 32 + lane) * 16);
            acc = wmma16(a, b, acc);
        }
        if (mat == 2) {                     // transposed store: Vt[d][m]
#pragma unroll
            for (int r = 0; r < 8; ++r)
                Vt[colg * VTS + mt * 16 + r + hi8] = (_Float16)acc[r];
        } else {
            _Float16* D = (mat == 0) ? Qh : Kh;
#pragma unroll
            for (int r = 0; r < 8; ++r)
                D[(mt * 16 + r + hi8) * XS + colg] = (_Float16)acc[r];
        }
    }
    __syncthreads();

    // ---- stage 3: S = Q @ K^T per head (6 heads * 4x4 tiles, K=32 -> 1 wmma)
    // B-fragment of K^T == A-style row load of K (lane <-> key row, contiguous d)
    for (int t = wave; t < 96; t += 8) {
        int h = t / 16, tt = t % 16;
        int mt = tt / 4, nt = tt % 4;
        v8f acc = {};
        v16h a = frag_lds(Qh + (mt * 16 + l15) * XS + h * 32 + kb);
        v16h b = frag_lds(Kh + (nt * 16 + l15) * XS + h * 32 + kb);
        acc = wmma16(a, b, acc);
#pragma unroll
        for (int r = 0; r < 8; ++r)
            Sf[h * 64 * SST + (mt * 16 + r + hi8) * SST + nt * 16 + l15] = acc[r];
    }
    __syncthreads();

    // ---- stage 4: softmax rows (add rel-bias + shifted-window mask), f32 -> P f16
    for (int rr = tid; rr < HEADS * NTOK; rr += 256) {
        int h = rr / NTOK, n = rr % NTOK;
        float* Srow = Sf + h * 64 * SST + n * SST;
        const float* brow = biasP + (h * NTOK + n) * NTOK;
        const float* mrow = mask + ((size_t)win * NTOK + n) * NTOK;
        float mx = -1e30f;
        for (int m = 0; m < NTOK; ++m) {
            float v = Srow[m] + brow[m] + mrow[m];
            Srow[m] = v;
            mx = fmaxf(mx, v);
        }
        float sum = 0.0f;
        for (int m = 0; m < NTOK; ++m) {
            float e = __expf(Srow[m] - mx);
            Srow[m] = e;
            sum += e;
        }
        float inv = 1.0f / sum;
        _Float16* Prow = Ph + (h * 64 + n) * PST;
        for (int m = 0; m < NTOK; ++m) Prow[m] = (_Float16)(Srow[m] * inv);
        for (int m = NTOK; m < 64; ++m) Prow[m] = (_Float16)0.0f;  // pad cols
    }
    // zero P pad rows (n = 49..63) so stage 5 fragments read clean zeros
    for (int idx = tid; idx < HEADS * 15 * (64 / 4); idx += 256) {
        int h = idx / (15 * 16), rem = idx % (15 * 16);
        int n = NTOK + rem / 16, m4 = rem % 16;
        v4h z; z[0] = z[1] = z[2] = z[3] = (_Float16)0.0f;
        *(v4h*)(Ph + (h * 64 + n) * PST + m4 * 4) = z;
    }
    __syncthreads();

    // ---- stage 5: O = P @ V per head (6 heads * 4x2 tiles, K=64 -> 2 wmma)
    //      B operand from Vt: lane <-> column d, contiguous in m -> row load
    for (int t = wave; t < 48; t += 8) {
        int h = t / 8, tt = t % 8;
        int mt = tt / 2, nt = tt % 2;
        v8f acc = {};
#pragma unroll
        for (int kt = 0; kt < 2; ++kt) {
            v16h a = frag_lds(Ph + (h * 64 + mt * 16 + l15) * PST + kt * 32 + kb);
            v16h b = frag_lds(Vt + (h * 32 + nt * 16 + l15) * VTS + kt * 32 + kb);
            acc = wmma16(a, b, acc);
        }
#pragma unroll
        for (int r = 0; r < 8; ++r)
            Oh[(mt * 16 + r + hi8) * XS + h * 32 + nt * 16 + l15] = (_Float16)acc[r];
    }
    __syncthreads();

    // ---- stage 6: out = O @ Wp + bp  (4x12 tiles, K=192 -> 6 wmma), f32 store
    for (int t = wave; t < 48; t += 8) {
        int mt = t / 12, nt = t % 12;
        int colg = nt * 16 + l15;
        float b0 = bp[colg];
        v8f acc;
#pragma unroll
        for (int r = 0; r < 8; ++r) acc[r] = b0;
        int arow = mt * 16 + l15;
#pragma unroll
        for (int kt = 0; kt < 6; ++kt) {
            v16h a = frag_lds(Oh + arow * XS + kt * 32 + kb);
            v16h b = frag_glb(wts + (((3 * 6 + kt) * 12 + nt) * 32 + lane) * 16);
            acc = wmma16(a, b, acc);
        }
#pragma unroll
        for (int r = 0; r < 8; ++r) {
            int row = mt * 16 + r + hi8;
            if (row < NTOK)
                out[((size_t)blk * NTOK + row) * DIM + colg] = acc[r];
        }
    }
}

// ---------------------------------------------------------------------------
extern "C" void kernel_launch(void* const* d_in, const int* in_sizes, int n_in,
                              void* d_out, int out_size, void* d_ws, size_t ws_size,
                              hipStream_t stream) {
    const float* x        = (const float*)d_in[0];
    const float* mask     = (const float*)d_in[1];
    const float* Wq       = (const float*)d_in[2];
    const float* bq       = (const float*)d_in[3];
    const float* Wk       = (const float*)d_in[4];
    const float* bk       = (const float*)d_in[5];
    const float* Wv       = (const float*)d_in[6];
    const float* bv       = (const float*)d_in[7];
    const float* Wp       = (const float*)d_in[8];
    const float* bp       = (const float*)d_in[9];
    const float* rel_bias = (const float*)d_in[10];
    const int*   pos_idx  = (const int*)d_in[11];
    float* out = (float*)d_out;

    // workspace: fragment-major f16 weights (294912 B) then f32 bias table
    _Float16* wts   = (_Float16*)d_ws;
    float*    biasP = (float*)((char*)d_ws + 4 * DIM * DIM * sizeof(_Float16));

    wa_convert_weights<<<(4 * DIM * DIM + 255) / 256, 256, 0, stream>>>(Wq, Wk, Wv, Wp, wts);
    wa_build_bias<<<(HEADS * NTOK * NTOK + 255) / 256, 256, 0, stream>>>(rel_bias, pos_idx, biasP);

    wa_attn_kernel<<<8192, 256, LDS_BYTES, stream>>>(x, mask, bq, bk, bv, bp,
                                                     wts, biasP, out);
}
